// KnowledgeAwareAttention_5007931867771
// MI455X (gfx1250) — matrix operand order repaired and checked by reading, not compiled, not run on hardware
//
#include <hip/hip_runtime.h>

// KnowledgeAwareAttention on MI455X (gfx1250, wave32, WMMA).
//
// attn[i,j] = sum_d R[q[i,j],d] * x[i,d] * x[j,d]  -> softmax_j -> W @ X.
// The O(B^2*D) gather is eliminated: per row i,
//     E_i = Xh_f16 @ (R (.) x[i])^T        (N = 42 relations padded to 48)
// via v_wmma_f32_16x16x32_f16; attn[i,j] = E_i[j, q[i,j]] from the LDS tile.
// Folding x[i] into the (tiny, per-block) B operand makes the inner loop pure
// load + WMMA: no per-chunk f32 multiplies or f16 converts.

#define BATCH 1024
#define DIM   256
#define NREL  42
#define NRELP 48   // padded to 3 N-tiles of 16

typedef __attribute__((ext_vector_type(16))) _Float16 v16h;
typedef __attribute__((ext_vector_type(8)))  float    v8f;

union H8 { uint4 u; _Float16 h[8]; };

// ---------------------------------------------------------------------------
// Kernel 0: Xh[j][d] = (f16)x[j][d]  (A-frags, kernel 1)
//           XT[d][j] = (f16)x[j][d]  (B-frags, kernel 2)
// ---------------------------------------------------------------------------
__global__ __launch_bounds__(256)
void prep_f16_kernel(const float* __restrict__ x,
                     _Float16* __restrict__ Xh,
                     _Float16* __restrict__ XT)
{
    int idx = blockIdx.x * 256 + threadIdx.x;      // idx = j*DIM + d
    int j = idx >> 8;
    int d = idx & (DIM - 1);
    float v = x[idx];
    Xh[idx] = (_Float16)v;
    XT[d * BATCH + j] = (_Float16)v;
}

// ---------------------------------------------------------------------------
// Kernel 1: one block per row i. E-tile GEMM (WMMA) + lookup + softmax -> W row
// ---------------------------------------------------------------------------
__global__ __launch_bounds__(256)
void attn_softmax_kernel(const float*    __restrict__ x,
                         const _Float16* __restrict__ Xh,
                         const int*      __restrict__ q,
                         const float*    __restrict__ R,
                         _Float16*       __restrict__ W)
{
    __shared__ __attribute__((aligned(32))) _Float16 sRp[NRELP][DIM]; // 24 KB: R'(n,k)=R(n,k)*x[i,k]
    __shared__ float sE[8][16 * NRELP];   // per-wave 16x48 E tile, 24 KB
    __shared__ float sRow[BATCH];         // attn row -> exp row
    __shared__ float sRed[8];

    const int i    = blockIdx.x;
    const int tid  = threadIdx.x;
    const int wave = tid >> 5;
    const int lane = tid & 31;

    // build scaled relation table R' = R (.) x[i]  (f16, zero-padded rows 42..47)
    const float* u = x + (size_t)i * DIM;
    for (int idx = tid; idx < NRELP * DIM; idx += 256) {
        int rel = idx / DIM, d = idx & (DIM - 1);
        float v = (rel < NREL) ? R[rel * DIM + d] * u[d] : 0.0f;
        sRp[rel][d] = (_Float16)v;
    }
    __syncthreads();

    const int m    = lane & 15;        // A-row / B-column / C-column index
    const int hi   = lane >> 4;        // lane half
    const int offA = hi ? 16 : 0;      // A: K contiguous 16 per lane half
    const int offB = hi ? 16 : 0;      // B: K contiguous 16 per lane half

    for (int jt = wave; jt < BATCH / 16; jt += 8) {
        const int j0 = jt * 16;
        v8f c0 = {}, c1 = {}, c2 = {};
        const _Float16* xr = Xh + (size_t)(j0 + m) * DIM;

        for (int kc = 0; kc < DIM / 32; ++kc) {
            const int kb = kc * 32;
            // A fragment: A[m,k] = Xh[j0+m, k] -- 16 contiguous halves (two 16B loads)
            H8 a0, a1;
            a0.u = *(const uint4*)(xr + kb + offA);
            a1.u = *(const uint4*)(xr + kb + offA + 8);
            v16h a;
#pragma unroll
            for (int e = 0; e < 8; ++e) { a[e] = a0.h[e]; a[e + 8] = a1.h[e]; }
            // B fragments: B[k,n] = R'[n0+n, k] (f16 LDS, 32B contiguous)
            v16h b0 = *(const v16h*)&sRp[m][kb + offB];
            v16h b1 = *(const v16h*)&sRp[16 + m][kb + offB];
            v16h b2 = *(const v16h*)&sRp[32 + m][kb + offB];

            c0 = __builtin_amdgcn_wmma_f32_16x16x32_f16(false, a, false, b0, (short)0, c0, false, false);
            c1 = __builtin_amdgcn_wmma_f32_16x16x32_f16(false, a, false, b1, (short)0, c1, false, false);
            c2 = __builtin_amdgcn_wmma_f32_16x16x32_f16(false, a, false, b2, (short)0, c2, false, false);
        }

        // spill E tile (16 x 48) to this wave's LDS scratch
        float* Ew = sE[wave];
#pragma unroll
        for (int r = 0; r < 8; ++r) {
            int row = r + (hi << 3);
            Ew[row * NRELP + m]      = c0[r];
            Ew[row * NRELP + 16 + m] = c1[r];
            Ew[row * NRELP + 32 + m] = c2[r];
        }
        // gather attn[i, j0+l] = E[l, q[i, j0+l]]  (same-wave DS ops are in-order)
        if (lane < 16) {
            int j   = j0 + lane;
            int rel = q[i * BATCH + j];
            sRow[j] = Ew[lane * NRELP + rel];
        }
    }
    __syncthreads();

    // ---- row softmax over 1024 entries ----
    float mx = -1e30f;
    for (int jj = tid; jj < BATCH; jj += 256) mx = fmaxf(mx, sRow[jj]);
#pragma unroll
    for (int s = 16; s > 0; s >>= 1) mx = fmaxf(mx, __shfl_xor(mx, s, 32));
    if (lane == 0) sRed[wave] = mx;
    __syncthreads();
    if (tid == 0) {
        float v = sRed[0];
        for (int w = 1; w < 8; ++w) v = fmaxf(v, sRed[w]);
        sRed[0] = v;
    }
    __syncthreads();
    const float rmax = sRed[0];

    float sum = 0.0f;
    for (int jj = tid; jj < BATCH; jj += 256) {
        float e = __expf(sRow[jj] - rmax);
        sRow[jj] = e;
        sum += e;
    }
#pragma unroll
    for (int s = 16; s > 0; s >>= 1) sum += __shfl_xor(sum, s, 32);
    __syncthreads();                       // everyone has read sRed[0]
    if (lane == 0) sRed[wave] = sum;
    __syncthreads();
    if (tid == 0) {
        float v = 0.0f;
        for (int w = 0; w < 8; ++w) v += sRed[w];
        sRed[0] = v;
    }
    __syncthreads();
    const float inv = 1.0f / sRed[0];

    _Float16* Wr = W + (size_t)i * BATCH;
    for (int jj = tid; jj < BATCH; jj += 256)
        Wr[jj] = (_Float16)(sRow[jj] * inv);
}

// ---------------------------------------------------------------------------
// Kernel 2: out = W (f16) @ X, one wave per 16x16 tile, K = 1024 in 32-chunks
// ---------------------------------------------------------------------------
__global__ __launch_bounds__(256)
void out_gemm_kernel(const _Float16* __restrict__ W,
                     const _Float16* __restrict__ XT,
                     float* __restrict__ out)
{
    const int tid  = threadIdx.x;
    const int wave = tid >> 5;
    const int lane = tid & 31;
    const int g  = blockIdx.x * 8 + wave;  // 0..1023 tiles
    const int i0 = (g >> 4) * 16;          // 64 i-tiles
    const int d0 = (g & 15) * 16;          // 16 d-tiles
    const int m    = lane & 15;
    const int hi   = lane >> 4;
    const int offA = hi ? 16 : 0;
    const int offB = hi ? 16 : 0;

    v8f c = {};
    const _Float16* Wr = W + (size_t)(i0 + m) * BATCH;
    for (int kc = 0; kc < BATCH / 32; ++kc) {
        const int kb = kc * 32;
        // A[m,k] = W[i0+m, k] : 16 contiguous halves (two 16B loads)
        H8 a0, a1;
        a0.u = *(const uint4*)(Wr + kb + offA);
        a1.u = *(const uint4*)(Wr + kb + offA + 8);
        v16h a;
#pragma unroll
        for (int e = 0; e < 8; ++e) { a[e] = a0.h[e]; a[e + 8] = a1.h[e]; }
        // B[k,n] = x[k, d0+n] = XT[d0+n][k] : 16 contiguous halves (32B)
        v16h b = *(const v16h*)(XT + (size_t)(d0 + m) * BATCH + kb + offB);

        c = __builtin_amdgcn_wmma_f32_16x16x32_f16(false, a, false, b, (short)0, c, false, false);
    }
    // C layout: VGPR r -> row (hi?8:0)+r, column lane&15
    float* op = out + (size_t)(i0 + (hi << 3)) * DIM + d0 + m;
#pragma unroll
    for (int r = 0; r < 8; ++r) op[r * DIM] = c[r];
}

// ---------------------------------------------------------------------------
extern "C" void kernel_launch(void* const* d_in, const int* in_sizes, int n_in,
                              void* d_out, int out_size, void* d_ws, size_t ws_size,
                              hipStream_t stream)
{
    const float* x = (const float*)d_in[0];   // [1024,256] f32
    const int*   q = (const int*)d_in[2];     // [1024,1024] int32
    const float* R = (const float*)d_in[4];   // [42,256] f32
    float* out = (float*)d_out;               // [1024,256] f32

    // workspace: W (f16, 2 MB) | XT (f16, 512 KB) | Xh (f16, 512 KB)
    _Float16* W  = (_Float16*)d_ws;
    _Float16* XT = (_Float16*)((char*)d_ws + (size_t)BATCH * BATCH * sizeof(_Float16));
    _Float16* Xh = (_Float16*)((char*)XT + (size_t)BATCH * DIM * sizeof(_Float16));

    prep_f16_kernel<<<(BATCH * DIM) / 256, 256, 0, stream>>>(x, Xh, XT);
    attn_softmax_kernel<<<BATCH, 256, 0, stream>>>(x, Xh, q, R, W);
    out_gemm_kernel<<<(BATCH / 16) * (DIM / 16) / 8, 256, 0, stream>>>(W, XT, out);
}
